// MultiViewEncoder_59313498357930
// MI455X (gfx1250) — compile-verified, matrix operand order checked by reading, not run on hardware
//
#include <hip/hip_runtime.h>
#include <hip/hip_bf16.h>

typedef _Float16 h16;
typedef __attribute__((ext_vector_type(16))) _Float16 v16h;
typedef __attribute__((ext_vector_type(8)))  _Float16 v8h;
typedef __attribute__((ext_vector_type(8)))  float    v8f;

__device__ __forceinline__ v8f zero8f() {
  v8f a;
#pragma unroll
  for (int i = 0; i < 8; ++i) a[i] = 0.f;
  return a;
}

__device__ __forceinline__ v8f wmma_f16(v16h a, v16h b, v8f c) {
  // (neg_a, A, neg_b, B, c_mod, C, reuse_a, reuse_b) -> v_wmma_f32_16x16x32_f16
  return __builtin_amdgcn_wmma_f32_16x16x32_f16(false, a, false, b, (short)0, c,
                                                false, false);
}

// A/B fragment loader for a 16-bit row-major tile (row stride ldh halves).
// ISA 16-bit A layout: lanes 0-15 -> rows 0-15 K-chunks {k0..k0+7, k0+16..k0+23},
// lanes 16-31 -> same rows, chunks shifted by 8. B uses the identical layout on W^T.
__device__ __forceinline__ v16h load_frag_h16(const h16* base, int ldh, int lane, int k0) {
  const int row = lane & 15;
  const int kb  = k0 + ((lane >> 4) << 3);
  const h16* p  = base + row * ldh + kb;
  v8h lo = *(const v8h*)(p);
  v8h hi = *(const v8h*)(p + 16);
  v16h a;
#pragma unroll
  for (int i = 0; i < 8; ++i) { a[i] = lo[i]; a[8 + i] = hi[i]; }
  return a;
}

// A fragment straight from f32 x[N,256], converted to f16, zero-filled past N.
__device__ __forceinline__ v16h load_frag_x(const float* x, int N, int r0, int lane, int k0) {
  const int row = r0 + (lane & 15);
  const int kb  = k0 + ((lane >> 4) << 3);
  v16h a;
  if (row < N) {
    const float* p = x + (size_t)row * 256 + kb;
#pragma unroll
    for (int i = 0; i < 8; ++i) { a[i] = (h16)p[i]; a[8 + i] = (h16)p[16 + i]; }
  } else {
#pragma unroll
    for (int i = 0; i < 16; ++i) a[i] = (h16)0;
  }
  return a;
}

// ---------------- weight convert + transpose: dst[n*K+k] = (f16)src[k*Nc+n] ----
__global__ void mv_cvt_t_kernel(const float* __restrict__ src, h16* __restrict__ dst,
                                int K, int Nc) {
  int n = blockIdx.x * 16 + threadIdx.x;
  int k = blockIdx.y * 16 + threadIdx.y;
  if (n < Nc && k < K) dst[(size_t)n * K + k] = (h16)src[(size_t)k * Nc + n];
}

// ---------------- exclusive prefix sum of n_views (G = 32768) ------------------
__global__ __launch_bounds__(1024) void mv_scan_kernel(const int* __restrict__ nv,
                                                       int* __restrict__ offs) {
  __shared__ int buf[1024];
  __shared__ int carry_s;
  const int tid = threadIdx.x;
  if (tid == 0) carry_s = 0;
  __syncthreads();
  for (int chunk = 0; chunk < 32; ++chunk) {
    const int idx = (chunk << 10) + tid;
    const int v = nv[idx];
    buf[tid] = v;
    __syncthreads();
    int acc = v;
    for (int off = 1; off < 1024; off <<= 1) {
      int t = (tid >= off) ? buf[tid - off] : 0;
      __syncthreads();
      acc += t;
      buf[tid] = acc;
      __syncthreads();
    }
    offs[idx] = carry_s + acc - v;   // exclusive
    __syncthreads();
    if (tid == 1023) carry_s += acc; // visible next chunk (barriers in between)
    __syncthreads();
  }
}

// ---------------- fused per-view pipeline: enc -> p1(tanh) -> e -> v -----------
// one wave per 16 view rows
__global__ __launch_bounds__(32) void mv_view_kernel(
    const float* __restrict__ x, const float* __restrict__ enc_b,
    const float* __restrict__ p1_b, const float* __restrict__ gat_a,
    const h16* __restrict__ encWt, const h16* __restrict__ p1Wt,
    const h16* __restrict__ gatWt, h16* __restrict__ h_out,
    float* __restrict__ e_out, h16* __restrict__ v_out, int N) {
  constexpr int LD0 = 136;   // 128 + 8 halves pad (keeps 16B align, staggers banks)
  constexpr int LD1 = 264;   // 256 + 8
  __shared__ h16 h0[16 * LD0];
  __shared__ h16 hl[16 * LD1];
  const int lane  = threadIdx.x;
  const int r0    = blockIdx.x * 16;
  const int colL  = lane & 15;
  const int rbase = (lane >> 4) << 3;   // D-frag: VGPR r -> row rbase+r, col colL

  // ---- stage 1: h0 = x @ enc_W + enc_b  (K=256 -> N=128) ----
  v16h ax[8];
#pragma unroll
  for (int k = 0; k < 8; ++k) ax[k] = load_frag_x(x, N, r0, lane, k * 32);
#pragma unroll 1
  for (int n = 0; n < 8; ++n) {
    v8f acc = zero8f();
#pragma unroll
    for (int k = 0; k < 8; ++k)
      acc = wmma_f16(ax[k], load_frag_h16(encWt + n * 16 * 256, 256, lane, k * 32), acc);
    const float bias = enc_b[n * 16 + colL];
#pragma unroll
    for (int r = 0; r < 8; ++r)
      h0[(rbase + r) * LD0 + n * 16 + colL] = (h16)(acc[r] + bias);
  }
  __syncthreads();

  // ---- stage 2: h = tanh(h0 @ p1_W + p1_b)  (K=128 -> N=256) ----
  v16h ah[4];
#pragma unroll
  for (int k = 0; k < 4; ++k) ah[k] = load_frag_h16(h0, LD0, lane, k * 32);
#pragma unroll 1
  for (int n = 0; n < 16; ++n) {
    v8f acc = zero8f();
#pragma unroll
    for (int k = 0; k < 4; ++k)
      acc = wmma_f16(ah[k], load_frag_h16(p1Wt + n * 16 * 128, 128, lane, k * 32), acc);
    const float bias = p1_b[n * 16 + colL];
#pragma unroll
    for (int r = 0; r < 8; ++r) {
      const float val = tanhf(acc[r] + bias);
      hl[(rbase + r) * LD1 + n * 16 + colL] = (h16)val;
      const int grow = r0 + rbase + r;
      if (grow < N) h_out[(size_t)grow * 256 + n * 16 + colL] = (h16)val;
    }
  }
  __syncthreads();

  // ---- stage 3: e = h @ gat_a ----
  {
    const int row = lane & 15, hf = lane >> 4;
    float s = 0.f;
    for (int c = hf * 128; c < hf * 128 + 128; ++c)
      s += (float)hl[row * LD1 + c] * gat_a[c];
    s += __shfl_xor(s, 16, 32);
    const int grow = r0 + row;
    if (hf == 0 && grow < N) e_out[grow] = s;
  }

  // ---- stage 4: v = h @ gat_W  (K=256 -> N=256) ----
  v16h av[8];
#pragma unroll
  for (int k = 0; k < 8; ++k) av[k] = load_frag_h16(hl, LD1, lane, k * 32);
#pragma unroll 1
  for (int n = 0; n < 16; ++n) {
    v8f acc = zero8f();
#pragma unroll
    for (int k = 0; k < 8; ++k)
      acc = wmma_f16(av[k], load_frag_h16(gatWt + n * 16 * 256, 256, lane, k * 32), acc);
#pragma unroll
    for (int r = 0; r < 8; ++r) {
      const int grow = r0 + rbase + r;
      if (grow < N) v_out[(size_t)grow * 256 + n * 16 + colL] = (h16)acc[r];
    }
  }
}

// ---------------- per-object: softmax over views, ctx, residual, max-pool ------
// one wave per object (cnt <= 8 contiguous rows), 8 cols per lane
__global__ __launch_bounds__(256) void mv_agg_kernel(
    const int* __restrict__ offs, const int* __restrict__ nv,
    const float* __restrict__ e, const h16* __restrict__ h,
    const h16* __restrict__ v, h16* __restrict__ pooled) {
  const int lane = threadIdx.x & 31;
  const int g = blockIdx.x * 8 + (threadIdx.x >> 5);
  const int r0 = offs[g];
  const int cnt = nv[g];
  float ev = (lane < cnt) ? e[r0 + lane] : -__builtin_inff();
  float m = ev;
#pragma unroll
  for (int off = 16; off > 0; off >>= 1) m = fmaxf(m, __shfl_xor(m, off, 32));
  const float w = (lane < cnt) ? __expf(ev - m) : 0.f;
  float den = w;
#pragma unroll
  for (int off = 16; off > 0; off >>= 1) den += __shfl_xor(den, off, 32);

  float ctx[8];
#pragma unroll
  for (int i = 0; i < 8; ++i) ctx[i] = 0.f;
  const h16* vb = v + (size_t)r0 * 256 + lane * 8;
  for (int c = 0; c < cnt; ++c) {
    const float a = __shfl(w, c, 32) / den;
    v8h vv = *(const v8h*)(vb + c * 256);
#pragma unroll
    for (int i = 0; i < 8; ++i) ctx[i] += a * (float)vv[i];
  }
  float pool[8];
#pragma unroll
  for (int i = 0; i < 8; ++i) pool[i] = -__builtin_inff();
  const h16* hb = h + (size_t)r0 * 256 + lane * 8;
  for (int c = 0; c < cnt; ++c) {
    v8h hv = *(const v8h*)(hb + c * 256);
#pragma unroll
    for (int i = 0; i < 8; ++i) pool[i] = fmaxf(pool[i], (float)hv[i] + ctx[i]);
  }
  v8h outv;
#pragma unroll
  for (int i = 0; i < 8; ++i) outv[i] = (h16)pool[i];
  *(v8h*)(pooled + (size_t)g * 256 + lane * 8) = outv;
}

// ---------------- head: tanh(pooled @ p2_W + b), L2 normalize ------------------
// one wave per 16 objects (G % 16 == 0)
__global__ __launch_bounds__(32) void mv_head_kernel(
    const h16* __restrict__ pooled, const h16* __restrict__ p2Wt,
    const float* __restrict__ p2_b, float* __restrict__ out) {
  constexpr int LDo = 132;
  __shared__ float ob[16 * LDo];
  const int lane = threadIdx.x;
  const int g0 = blockIdx.x * 16;
  const int colL = lane & 15;
  const int rbase = (lane >> 4) << 3;
  v16h ap[8];
#pragma unroll
  for (int k = 0; k < 8; ++k)
    ap[k] = load_frag_h16(pooled + (size_t)g0 * 256, 256, lane, k * 32);
#pragma unroll 1
  for (int n = 0; n < 8; ++n) {
    v8f acc = zero8f();
#pragma unroll
    for (int k = 0; k < 8; ++k)
      acc = wmma_f16(ap[k], load_frag_h16(p2Wt + n * 16 * 256, 256, lane, k * 32), acc);
    const float bias = p2_b[n * 16 + colL];
#pragma unroll
    for (int r = 0; r < 8; ++r)
      ob[(rbase + r) * LDo + n * 16 + colL] = tanhf(acc[r] + bias);
  }
  __syncthreads();
  const int row = lane & 15, hf = lane >> 4;
  float s = 0.f;
  for (int c = hf * 64; c < hf * 64 + 64; ++c) { float t = ob[row * LDo + c]; s += t * t; }
  s += __shfl_xor(s, 16, 32);
  const float scale = 1.f / (sqrtf(s) + 1e-9f);
  for (int c = hf * 64; c < hf * 64 + 64; ++c)
    out[(size_t)(g0 + row) * 128 + c] = ob[row * LDo + c] * scale;
}

extern "C" void kernel_launch(void* const* d_in, const int* in_sizes, int n_in,
                              void* d_out, int out_size, void* d_ws, size_t ws_size,
                              hipStream_t stream) {
  const float* x     = (const float*)d_in[0];
  const float* enc_W = (const float*)d_in[1];
  const float* enc_b = (const float*)d_in[2];
  const float* p1_W  = (const float*)d_in[3];
  const float* p1_b  = (const float*)d_in[4];
  const float* gat_a = (const float*)d_in[5];
  const float* gat_W = (const float*)d_in[6];
  const float* p2_W  = (const float*)d_in[7];
  const float* p2_b  = (const float*)d_in[8];
  const int* n_views = (const int*)d_in[9];

  const int N = in_sizes[0] / 256;
  constexpr int G = 32768;

  auto aup = [](size_t v) { return (v + 255) & ~(size_t)255; };
  char* ws = (char*)d_ws;
  size_t off = 0;
  h16* encWt  = (h16*)(ws + off); off = aup(off + (size_t)128 * 256 * sizeof(h16));
  h16* p1Wt   = (h16*)(ws + off); off = aup(off + (size_t)256 * 128 * sizeof(h16));
  h16* gatWt  = (h16*)(ws + off); off = aup(off + (size_t)256 * 256 * sizeof(h16));
  h16* p2Wt   = (h16*)(ws + off); off = aup(off + (size_t)128 * 256 * sizeof(h16));
  int*  offs  = (int*)(ws + off); off = aup(off + (size_t)G * sizeof(int));
  float* e_ws = (float*)(ws + off); off = aup(off + (size_t)N * sizeof(float));
  h16* h_ws   = (h16*)(ws + off); off = aup(off + (size_t)N * 256 * sizeof(h16));
  h16* v_ws   = (h16*)(ws + off); off = aup(off + (size_t)N * 256 * sizeof(h16));
  h16* pooled = (h16*)(ws + off); off = aup(off + (size_t)G * 256 * sizeof(h16));

  dim3 b16(16, 16);
  mv_cvt_t_kernel<<<dim3(128 / 16, 256 / 16), b16, 0, stream>>>(enc_W, encWt, 256, 128);
  mv_cvt_t_kernel<<<dim3(256 / 16, 128 / 16), b16, 0, stream>>>(p1_W,  p1Wt,  128, 256);
  mv_cvt_t_kernel<<<dim3(256 / 16, 256 / 16), b16, 0, stream>>>(gat_W, gatWt, 256, 256);
  mv_cvt_t_kernel<<<dim3(128 / 16, 256 / 16), b16, 0, stream>>>(p2_W,  p2Wt,  256, 128);

  mv_scan_kernel<<<1, 1024, 0, stream>>>(n_views, offs);

  const int tiles = (N + 15) / 16;
  mv_view_kernel<<<tiles, 32, 0, stream>>>(x, enc_b, p1_b, gat_a, encWt, p1Wt,
                                           gatWt, h_ws, e_ws, v_ws, N);

  mv_agg_kernel<<<G / 8, 256, 0, stream>>>(offs, n_views, e_ws, h_ws, v_ws, pooled);

  mv_head_kernel<<<G / 16, 32, 0, stream>>>(pooled, p2Wt, p2_b, (float*)d_out);
}
